// LDGNNetwork_67886253081260
// MI455X (gfx1250) — matrix operand order repaired and checked by reading, not compiled, not run on hardware
//
#include <hip/hip_runtime.h>

typedef __attribute__((ext_vector_type(16))) _Float16 v16h;
typedef __attribute__((ext_vector_type(8)))  _Float16 v8h;
typedef __attribute__((ext_vector_type(8)))  float    v8f;

#define NEG_SLOPE 0.2f

// ---------- float <-> orderable uint (for atomic float max) ----------
__device__ __forceinline__ unsigned f2ord(float f) {
  unsigned u = __float_as_uint(f);
  return (u & 0x80000000u) ? ~u : (u | 0x80000000u);
}
__device__ __forceinline__ float ord2f(unsigned u) {
  return __uint_as_float((u & 0x80000000u) ? (u & 0x7FFFFFFFu) : ~u);
}
// pack two f32 -> packed 2x f16 in a u32
__device__ __forceinline__ unsigned pack2(float a, float b) {
  union { _Float16 h; unsigned short u; } ua, ub;
  ua.h = (_Float16)a; ub.h = (_Float16)b;
  return (unsigned)ua.u | ((unsigned)ub.u << 16);
}

// ---------- WMMA fragment loaders (ISA 7.12.2 layouts, wave32) ----------
// A (16x32 f16): lane L holds row L%16; K = khi..khi+7 and 16+khi..+23, khi=(L>=16)?8:0
// p points at (row, kbase+khi); halves [0..7] and [16..23] -> two 16B vector loads.
__device__ __forceinline__ v16h load_a16(const _Float16* p) {
  const v8h* q = (const v8h*)p;
  union { v16h v; v8h h[2]; } u;
  u.h[0] = q[0];
  u.h[1] = q[2];
  return u.v;
}
// B fragment from fragment-major swizzled LDS: lane's 16 halves contiguous (32B)
__device__ __forceinline__ v16h load_bfrag(const _Float16* base) {
  const v8h* q = (const v8h*)base;
  union { v16h v; v8h h[2]; } u;
  u.h[0] = q[0];
  u.h[1] = q[1];
  return u.v;
}

// =====================================================================
// Kernel 1: fused encoder MLP  x_enc = relu(relu(x@W1+b1)@W2+b2)
// 8 waves/block, each wave owns a 16-row tile.  WMMA f16 -> f32.
// Weights pre-swizzled in LDS to fragment-major order.
// =====================================================================
__global__ __launch_bounds__(256) void enc_kernel(
    const float* __restrict__ x_obs, const float* __restrict__ w1,
    const float* __restrict__ b1, const float* __restrict__ w2,
    const float* __restrict__ b2, float* __restrict__ x_enc,
    _Float16* __restrict__ x_enc_h, int N) {
  __shared__ __align__(16) _Float16 w1f[4 * 512];   // frag(ct) for [32x64], K padded 16->32
  __shared__ __align__(16) _Float16 w2f[8 * 512];   // frag(ct*2+ks) for [64x64]
  __shared__ float b1s[64], b2s[64];
  __shared__ __align__(16) _Float16 buf[8][16 * 72]; // per-wave tile, padded stride

  __builtin_prefetch(w2, 0, 1);

  // swizzled fill: fragment f, lane lw, pair j -> W[k..k+1][col]
  unsigned* w1u = (unsigned*)w1f;
  for (int i = threadIdx.x; i < 4 * 256; i += 256) {
    int h2 = i * 2;
    int frag = h2 >> 9, within = h2 & 511;
    int lw = within >> 4, j = within & 15;
    int col = frag * 16 + (lw & 15);
    int k = ((j >> 3) << 4) + ((lw >> 4) << 3) + (j & 7);
    float a = (k < 16) ? w1[k * 64 + col] : 0.f;
    float bb = (k + 1 < 16) ? w1[(k + 1) * 64 + col] : 0.f;
    w1u[i] = pack2(a, bb);
  }
  unsigned* w2u = (unsigned*)w2f;
  for (int i = threadIdx.x; i < 8 * 256; i += 256) {
    int h2 = i * 2;
    int frag = h2 >> 9, within = h2 & 511;
    int lw = within >> 4, j = within & 15;
    int ct = frag >> 1, ks = frag & 1;
    int col = ct * 16 + (lw & 15);
    int k = ks * 32 + ((j >> 3) << 4) + ((lw >> 4) << 3) + (j & 7);
    w2u[i] = pack2(w2[k * 64 + col], w2[(k + 1) * 64 + col]);
  }
  if (threadIdx.x < 64) { b1s[threadIdx.x] = b1[threadIdx.x]; b2s[threadIdx.x] = b2[threadIdx.x]; }
  __syncthreads();

  int lane = threadIdx.x & 31;
  int wave = threadIdx.x >> 5;
  int tile = blockIdx.x * 8 + wave;
  if (tile * 16 >= N) return;
  int r0 = tile * 16;
  _Float16* mybuf = buf[wave];

  // stage x_obs rows (K=16, pad to 32 with zeros), packed b32 stores
  unsigned* bu = (unsigned*)mybuf;
  for (int i = lane; i < 16 * 16; i += 32) {
    int r = i >> 4, cu = i & 15;                 // uint column (2 halves)
    unsigned val = 0;
    if (cu < 8) val = pack2(x_obs[(r0 + r) * 16 + cu * 2], x_obs[(r0 + r) * 16 + cu * 2 + 1]);
    bu[r * 36 + cu] = val;
  }

  int n = lane & 15;
  int mb = (lane >> 4) << 3;
  int khi = (lane >> 4) << 3;

  // layer 1: [16x32] x [32x64]
  v16h a = load_a16(mybuf + (lane & 15) * 72 + khi);
  v8f h1acc[4];
#pragma unroll
  for (int ct = 0; ct < 4; ++ct) {
    v8f c0 = {};
    c0 = __builtin_amdgcn_wmma_f32_16x16x32_f16(
        false, a, false, load_bfrag(w1f + ct * 512 + lane * 16),
        (short)0, c0, false, false);
#pragma unroll
    for (int i = 0; i < 8; ++i) {
      float v = c0[i] + b1s[ct * 16 + n];
      h1acc[ct][i] = v > 0.f ? v : 0.f;
    }
  }
  // write h1 back to LDS tile (f16)
#pragma unroll
  for (int ct = 0; ct < 4; ++ct)
#pragma unroll
    for (int i = 0; i < 8; ++i)
      mybuf[(mb + i) * 72 + ct * 16 + n] = (_Float16)h1acc[ct][i];

  // layer 2: [16x64] x [64x64]
  v16h a0 = load_a16(mybuf + (lane & 15) * 72 + khi);
  v16h a1 = load_a16(mybuf + (lane & 15) * 72 + 32 + khi);
#pragma unroll
  for (int ct = 0; ct < 4; ++ct) {
    v8f acc = {};
    acc = __builtin_amdgcn_wmma_f32_16x16x32_f16(
        false, a0, false, load_bfrag(w2f + (ct * 2 + 0) * 512 + lane * 16), (short)0, acc, false, false);
    acc = __builtin_amdgcn_wmma_f32_16x16x32_f16(
        false, a1, false, load_bfrag(w2f + (ct * 2 + 1) * 512 + lane * 16), (short)0, acc, false, false);
#pragma unroll
    for (int i = 0; i < 8; ++i) {
      float v = acc[i] + b2s[ct * 16 + n];
      v = v > 0.f ? v : 0.f;
      x_enc[(size_t)(r0 + mb + i) * 64 + ct * 16 + n] = v;
      x_enc_h[(size_t)(r0 + mb + i) * 64 + ct * 16 + n] = (_Float16)v;
    }
  }
}

// =====================================================================
// Kernel 2: generic node GEMM  out[N,128] = A[N,K](f16) @ W[K,128]
// Weights swizzled fragment-major in LDS; A fragments via b128 global loads.
// =====================================================================
template <int K>
__global__ __launch_bounds__(256) void node_gemm_kernel(
    const _Float16* __restrict__ A, const float* __restrict__ W,
    float* __restrict__ out, int N) {
  constexpr int KS = K / 32;
  __shared__ __align__(16) _Float16 ws[K * 128];

  __builtin_prefetch(W + threadIdx.x * 16, 0, 1);

  unsigned* wsu = (unsigned*)ws;
  for (int i = threadIdx.x; i < K * 64; i += 256) {
    int h2 = i * 2;
    int frag = h2 >> 9, within = h2 & 511;
    int lw = within >> 4, j = within & 15;
    int ct = frag / KS, ks = frag % KS;
    int col = ct * 16 + (lw & 15);
    int k = ks * 32 + ((j >> 3) << 4) + ((lw >> 4) << 3) + (j & 7);
    wsu[i] = pack2(W[k * 128 + col], W[(k + 1) * 128 + col]);
  }
  __syncthreads();

  int lane = threadIdx.x & 31;
  int wave = threadIdx.x >> 5;
  int tiles = N >> 4;
  int n = lane & 15;
  int mb = (lane >> 4) << 3;
  int khi = (lane >> 4) << 3;

  for (int tile = blockIdx.x * 8 + wave; tile < tiles; tile += gridDim.x * 8) {
    int r0 = tile << 4;
    const _Float16* rowbase = A + (size_t)(r0 + (lane & 15)) * K;
    v16h afr[KS];
#pragma unroll
    for (int ks = 0; ks < KS; ++ks) afr[ks] = load_a16(rowbase + ks * 32 + khi);
#pragma unroll
    for (int ct = 0; ct < 8; ++ct) {
      v8f acc = {};
#pragma unroll
      for (int ks = 0; ks < KS; ++ks)
        acc = __builtin_amdgcn_wmma_f32_16x16x32_f16(
            false, afr[ks], false, load_bfrag(ws + (ct * KS + ks) * 512 + lane * 16),
            (short)0, acc, false, false);
#pragma unroll
      for (int i = 0; i < 8; ++i)
        out[(size_t)(r0 + mb + i) * 128 + ct * 16 + n] = acc[i];
    }
  }
}

// =====================================================================
// Edge kernels (GATv2 softmax over incoming edges, via atomics)
// =====================================================================
// wave per edge: coalesced float4 gathers, shuffle-reduce per 16-lane head
__global__ void edge_score_kernel(const float* __restrict__ xl, const float* __restrict__ xr,
                                  const float* __restrict__ att, const int* __restrict__ src,
                                  const int* __restrict__ dst, float* __restrict__ score,
                                  unsigned* __restrict__ mord, int E) {
  int lane = threadIdx.x & 31;
  int wid = (blockIdx.x * blockDim.x + threadIdx.x) >> 5;
  int nw = (gridDim.x * blockDim.x) >> 5;
  int c0 = lane * 4;
  float4 av = *(const float4*)(att + c0);
  for (int e = wid; e < E; e += nw) {
    int s = src[e], d = dst[e];
    float4 l = *(const float4*)(xl + (size_t)s * 128 + c0);
    float4 r = *(const float4*)(xr + (size_t)d * 128 + c0);
    float v, p;
    v = l.x + r.x; v = v > 0.f ? v : NEG_SLOPE * v; p  = v * av.x;
    v = l.y + r.y; v = v > 0.f ? v : NEG_SLOPE * v; p += v * av.y;
    v = l.z + r.z; v = v > 0.f ? v : NEG_SLOPE * v; p += v * av.z;
    v = l.w + r.w; v = v > 0.f ? v : NEG_SLOPE * v; p += v * av.w;
    // reduce within each 16-lane half (head 0: lanes 0-15, head 1: 16-31)
    p += __shfl_xor(p, 1, 32);
    p += __shfl_xor(p, 2, 32);
    p += __shfl_xor(p, 4, 32);
    p += __shfl_xor(p, 8, 32);
    if ((lane & 15) == 0) {
      int h = lane >> 4;
      score[e * 2 + h] = p;
      atomicMax(&mord[d * 2 + h], f2ord(p));
    }
  }
}

__global__ void edge_exp_kernel(const float* __restrict__ score, const unsigned* __restrict__ mord,
                                const int* __restrict__ dst, float* __restrict__ exb,
                                float* __restrict__ denom, int E) {
  for (int e = blockIdx.x * blockDim.x + threadIdx.x; e < E; e += gridDim.x * blockDim.x) {
    int d = dst[e];
    float2 sc = *(const float2*)(score + (size_t)e * 2);
    float ex0 = __expf(sc.x - ord2f(mord[d * 2 + 0]));
    float ex1 = __expf(sc.y - ord2f(mord[d * 2 + 1]));
    *(float2*)(exb + (size_t)e * 2) = make_float2(ex0, ex1);
    atomicAdd(&denom[d * 2 + 0], ex0);
    atomicAdd(&denom[d * 2 + 1], ex1);
  }
}

// one wave per edge: 32 lanes x 4 channels = 128 channels
__global__ void edge_agg_kernel(const float* __restrict__ exb, const float* __restrict__ denom,
                                const int* __restrict__ src, const int* __restrict__ dst,
                                const float* __restrict__ xl, float* __restrict__ agg, int E) {
  int lane = threadIdx.x & 31;
  int wid = (blockIdx.x * blockDim.x + threadIdx.x) >> 5;
  int nw = (gridDim.x * blockDim.x) >> 5;
  int c0 = lane * 4;
  int h = c0 >> 6;
  for (int e = wid; e < E; e += nw) {
    int s = src[e], d = dst[e];
    float alpha = exb[(size_t)e * 2 + h] / (denom[(size_t)d * 2 + h] + 1e-16f);
    const float4 v = *(const float4*)(xl + (size_t)s * 128 + c0);
    float* base = agg + (size_t)d * 128 + c0;
    atomicAdd(base + 0, alpha * v.x);
    atomicAdd(base + 1, alpha * v.y);
    atomicAdd(base + 2, alpha * v.z);
    atomicAdd(base + 3, alpha * v.w);
  }
}

// =====================================================================
// Elementwise / init / final kernels
// =====================================================================
__global__ void post1_kernel(const float* __restrict__ agg, const float* __restrict__ bias,
                             const float* __restrict__ dm, float* __restrict__ xrelu,
                             _Float16* __restrict__ xmh, int total) {
  for (int i = blockIdx.x * blockDim.x + threadIdx.x; i < total; i += gridDim.x * blockDim.x) {
    float v = agg[i] + bias[i & 127];
    v = v > 0.f ? v : 0.f;
    xrelu[i] = v;
    xmh[i] = (_Float16)(v * dm[i >> 7]);
  }
}

__global__ void init_kernel(float* __restrict__ agg, float* __restrict__ denom,
                            unsigned* __restrict__ mord, int n128, int n2) {
  for (int i = blockIdx.x * blockDim.x + threadIdx.x; i < n128; i += gridDim.x * blockDim.x) {
    agg[i] = 0.f;
    if (i < n2) { denom[i] = 0.f; mord[i] = 0x007FFFFFu; }  // f2ord(-inf)
  }
}

__global__ void final_kernel(const float* __restrict__ x_enc, const float* __restrict__ xrelu1,
                             const float* __restrict__ agg2, const float* __restrict__ bias2,
                             const float* __restrict__ out_w, const float* __restrict__ out_b,
                             const int* __restrict__ gidx, float* __restrict__ out, int BS) {
  int t = blockIdx.x * blockDim.x + threadIdx.x;
  if (t >= BS * 5) return;
  int row = t / 5, o = t % 5;
  int g = gidx[row];
  float acc = out_b[o];
  for (int f = 0; f < 64; ++f)
    acc += x_enc[(size_t)g * 64 + f] * out_w[f * 5 + o];
  for (int f = 0; f < 128; ++f)
    acc += xrelu1[(size_t)g * 128 + f] * out_w[(64 + f) * 5 + o];
  for (int f = 0; f < 128; ++f) {
    float v = agg2[(size_t)g * 128 + f] + bias2[f];
    v = v > 0.f ? v : 0.f;
    acc += v * out_w[(192 + f) * 5 + o];
  }
  out[t] = acc;
}

// =====================================================================
// Launch
// =====================================================================
extern "C" void kernel_launch(void* const* d_in, const int* in_sizes, int n_in,
                              void* d_out, int out_size, void* d_ws, size_t ws_size,
                              hipStream_t stream) {
  const float* x_obs = (const float*)d_in[0];
  const float* dm_mask = (const float*)d_in[1];
  const int*   edge_index = (const int*)d_in[2];
  const int*   gidx = (const int*)d_in[3];
  const float* enc_w1 = (const float*)d_in[4];
  const float* enc_b1 = (const float*)d_in[5];
  const float* enc_w2 = (const float*)d_in[6];
  const float* enc_b2 = (const float*)d_in[7];
  const float* wl1 = (const float*)d_in[8];
  const float* wr1 = (const float*)d_in[9];
  const float* att1 = (const float*)d_in[10];
  const float* bias1 = (const float*)d_in[11];
  const float* wl2 = (const float*)d_in[12];
  const float* wr2 = (const float*)d_in[13];
  const float* att2 = (const float*)d_in[14];
  const float* bias2 = (const float*)d_in[15];
  const float* out_w = (const float*)d_in[16];
  const float* out_b = (const float*)d_in[17];

  const int N = in_sizes[0] / 16;
  const int E = in_sizes[2] / 2;
  const int BS = in_sizes[3];
  const int* src = edge_index;
  const int* dst = edge_index + E;

  // workspace carve-out (256B aligned)
  char* w = (char*)d_ws;
  auto alloc = [&](size_t bytes) -> char* {
    char* p = w;
    w += (bytes + 255) & ~(size_t)255;
    return p;
  };
  float*    x_enc   = (float*)alloc((size_t)N * 64 * 4);
  _Float16* x_enc_h = (_Float16*)alloc((size_t)N * 64 * 2);
  float*    xl      = (float*)alloc((size_t)N * 128 * 4);
  float*    xr      = (float*)alloc((size_t)N * 128 * 4);
  float*    score   = (float*)alloc((size_t)E * 2 * 4);
  float*    exb     = (float*)alloc((size_t)E * 2 * 4);
  unsigned* mord    = (unsigned*)alloc((size_t)N * 2 * 4);
  float*    denom   = (float*)alloc((size_t)N * 2 * 4);
  float*    agg     = (float*)alloc((size_t)N * 128 * 4);
  float*    xrelu1  = (float*)alloc((size_t)N * 128 * 4);
  _Float16* xmh     = (_Float16*)alloc((size_t)N * 128 * 2);

  const int tiles = N / 16;
  const int gemm_blocks = (tiles + 7) / 8;
  const int n128 = N * 128;
  const int ew_blocks = (E + 255) / 256;            // thread-per-edge kernels
  const int wave_blocks = (E + 7) / 8;              // wave-per-edge kernels
  const int el_blocks = (n128 + 255) / 256;

  // --- encoder ---
  enc_kernel<<<gemm_blocks, 256, 0, stream>>>(x_obs, enc_w1, enc_b1, enc_w2, enc_b2,
                                              x_enc, x_enc_h, N);

  // --- layer 1 ---
  init_kernel<<<el_blocks, 256, 0, stream>>>(agg, denom, mord, n128, N * 2);
  node_gemm_kernel<64><<<gemm_blocks, 256, 0, stream>>>(x_enc_h, wl1, xl, N);
  node_gemm_kernel<64><<<gemm_blocks, 256, 0, stream>>>(x_enc_h, wr1, xr, N);
  edge_score_kernel<<<wave_blocks, 256, 0, stream>>>(xl, xr, att1, src, dst, score, mord, E);
  edge_exp_kernel<<<ew_blocks, 256, 0, stream>>>(score, mord, dst, exb, denom, E);
  edge_agg_kernel<<<wave_blocks, 256, 0, stream>>>(exb, denom, src, dst, xl, agg, E);
  post1_kernel<<<el_blocks, 256, 0, stream>>>(agg, bias1, dm_mask, xrelu1, xmh, n128);

  // --- layer 2 ---
  init_kernel<<<el_blocks, 256, 0, stream>>>(agg, denom, mord, n128, N * 2);
  node_gemm_kernel<128><<<gemm_blocks, 256, 0, stream>>>(xmh, wl2, xl, N);
  node_gemm_kernel<128><<<gemm_blocks, 256, 0, stream>>>(xmh, wr2, xr, N);
  edge_score_kernel<<<wave_blocks, 256, 0, stream>>>(xl, xr, att2, src, dst, score, mord, E);
  edge_exp_kernel<<<ew_blocks, 256, 0, stream>>>(score, mord, dst, exb, denom, E);
  edge_agg_kernel<<<wave_blocks, 256, 0, stream>>>(exb, denom, src, dst, xl, agg, E);

  // --- head ---
  final_kernel<<<(BS * 5 + 255) / 256, 256, 0, stream>>>(x_enc, xrelu1, agg, bias2,
                                                         out_w, out_b, gidx,
                                                         (float*)d_out, BS);
}
